// CIDER_25383256719536
// MI455X (gfx1250) — matrix operand order repaired and checked by reading, not compiled, not run on hardware
//
#include <hip/hip_runtime.h>

// ---------------------------------------------------------------------------
// Problem constants (match the reference)
// ---------------------------------------------------------------------------
constexpr int D_DIM   = 512;
constexpr int B_BATCH = 2048;
constexpr int S_SEQ   = 32;
constexpr int M_IND   = 16;
constexpr int N_HEADS = 8;
constexpr int HD      = 64;   // D / HEADS

// ---------------------------------------------------------------------------
// WMMA types (CDNA5 gfx1250, wave32)
// ---------------------------------------------------------------------------
typedef __attribute__((ext_vector_type(16))) __bf16 v16bf;
typedef __attribute__((ext_vector_type(8)))  float  v8f;

union FragBF { v16bf v; uint4 q[2]; };

__device__ __forceinline__ unsigned short f2bf(float f) {
  // round-to-nearest-even fp32 -> bf16
  unsigned int u = __float_as_uint(f);
  return (unsigned short)((u + 0x7FFFu + ((u >> 16) & 1u)) >> 16);
}

// ---------------------------------------------------------------------------
// One-time conversions (O(N^2), hoisted out of the O(N^3) GEMM loops)
// ---------------------------------------------------------------------------
// flat fp32 -> bf16
__global__ __launch_bounds__(256)
void cvt_bf16(const float* __restrict__ in, unsigned short* __restrict__ out) {
  const size_t i = (size_t)blockIdx.x * 256 + threadIdx.x;
  out[i] = f2bf(in[i]);
}

// inducing points -> bf16, padded to 64 rows with zeros (GEMM row-block = 64)
__global__ __launch_bounds__(256)
void prep_I_bf16(const float* __restrict__ I, unsigned short* __restrict__ out) {
  const int i = blockIdx.x * 256 + threadIdx.x;     // 64*512 elements
  const int row = i >> 9;
  out[i] = (row < M_IND) ? f2bf(I[i]) : (unsigned short)0;
}

// W[k][n] fp32 -> Wt[n][k] bf16 (coalesced writes; one-time, L2-resident after)
__global__ __launch_bounds__(256)
void wtrans_bf16(const float* __restrict__ W, unsigned short* __restrict__ Wt) {
  const int i = blockIdx.x * 256 + threadIdx.x;     // 512*512 elements
  const int n = i >> 9, k = i & 511;
  Wt[i] = f2bf(W[(size_t)k * D_DIM + n]);
}

// ---------------------------------------------------------------------------
// Fused GEMM: C[r, :] = A[r, :] @ W + bias   (K = N = 512), all-bf16 operands
// MODE 0: C = acc + bias
// MODE 1: C = Res + relu(acc + bias)         (the  O + relu(fc_o(O))  step)
//
// A  : bf16 row-major [nrows][512]    (LN shadow copies / pre-converted x)
// Wt : bf16 transposed [n][k]         -> B fragment = 2 aligned b128 loads
// Inner loop is pure global_load_b128 + v_wmma (no conversion, no LDS, no
// barriers). Block = 8 waves, block tile 64x256, wave tile 32x64 (2x4 WMMA).
// ---------------------------------------------------------------------------
template <int MODE>
__global__ __launch_bounds__(256)
void gemm_bf16_wmma(const unsigned short* __restrict__ Ab,
                    const unsigned short* __restrict__ Wt,
                    const float* __restrict__ bias, const float* __restrict__ Res,
                    float* __restrict__ C, int nrows)
{
  const int lane = threadIdx.x & 31;
  const int wid  = threadIdx.x >> 5;
  const int rowBlock = blockIdx.y * 64;
  const int colBlock = blockIdx.x * 256;
  const int waveRow  = (wid >> 2) * 32;   // 0 / 32
  const int waveCol  = (wid & 3) * 64;    // 0 / 64 / 128 / 192

  const int lnn = lane & 15;              // M (A) / N (B) within 16-wide tile
  const int kg  = lane >> 4;              // K lane-group
  const int lm  = kg * 8;                 // C/D: lanes 16-31 hold M+8

  // A fragment rows (bf16, row stride 512) and B fragment rows (Wt[n][k])
  const unsigned short* aRow0 = Ab + (size_t)(rowBlock + waveRow + lnn) * D_DIM;
  const unsigned short* aRow1 = aRow0 + (size_t)16 * D_DIM;
  const unsigned short* bRow  = Wt + (size_t)(colBlock + waveCol + lnn) * D_DIM;

  v8f acc[2][4];
#pragma unroll
  for (int i = 0; i < 2; ++i)
#pragma unroll
    for (int t = 0; t < 4; ++t) acc[i][t] = {};

  for (int ks = 0; ks < D_DIM; ks += 32) {
    if (ks + 32 < D_DIM) {  // gfx1250 global_prefetch_b8 of next K tile
      __builtin_prefetch(aRow0 + ks + 32, 0, 1);
      __builtin_prefetch(bRow + ks + 32, 0, 1);
    }
    // A (16x32 bf16): elems 0..7 -> K = ks+kg*8.., elems 8..15 -> K = ks+16+kg*8..
    FragBF a0, a1;
    a0.q[0] = *(const uint4*)(aRow0 + ks + kg * 8);
    a0.q[1] = *(const uint4*)(aRow0 + ks + 16 + kg * 8);
    a1.q[0] = *(const uint4*)(aRow1 + ks + kg * 8);
    a1.q[1] = *(const uint4*)(aRow1 + ks + 16 + kg * 8);
    // B (32x16 bf16): elems 0..7 -> K = ks+kg*8.., elems 8..15 -> K = ks+16+kg*8.., N = lane&15
#pragma unroll
    for (int t = 0; t < 4; ++t) {
      FragBF b;
      const unsigned short* br = bRow + (size_t)t * 16 * D_DIM + ks + kg * 8;
      b.q[0] = *(const uint4*)(br);
      b.q[1] = *(const uint4*)(br + 16);
      acc[0][t] = __builtin_amdgcn_wmma_f32_16x16x32_bf16(false, a0.v, false, b.v, (short)0, acc[0][t], false, false);
      acc[1][t] = __builtin_amdgcn_wmma_f32_16x16x32_bf16(false, a1.v, false, b.v, (short)0, acc[1][t], false, false);
    }
  }

  const int baseRow = rowBlock + waveRow;
  const int baseCol = colBlock + waveCol;
#pragma unroll
  for (int i = 0; i < 2; ++i) {
#pragma unroll
    for (int t = 0; t < 4; ++t) {
      const int col = baseCol + t * 16 + lnn;
      const float bv = bias[col];
#pragma unroll
      for (int g = 0; g < 8; ++g) {
        const int row = baseRow + i * 16 + lm + g;
        if (row < nrows) {
          const size_t idx = (size_t)row * D_DIM + col;
          float v = acc[i][t][g] + bv;
          if (MODE == 1) v = Res[idx] + fmaxf(v, 0.0f);
          C[idx] = v;
        }
      }
    }
  }
}

// ---------------------------------------------------------------------------
// Per-(batch, head) attention with residual on projected Q:
//   Oh = Qh + softmax(Qh Kh^T / sqrt(512)) Vh
// qStride == 0 -> Q shared across batches (mab0: projected inducing points).
// Tiny shapes (Sq,Sk <= 32, hd=64): HBM-bound, fp32 VALU + LDS.
// ---------------------------------------------------------------------------
__global__ __launch_bounds__(256)
void attn_512(const float* __restrict__ Qb, long qStride,
              const float* __restrict__ K, const float* __restrict__ V,
              float* __restrict__ Out, int Sq, int Sk)
{
  __shared__ float Qs[32 * HD];
  __shared__ float Ks[32 * HD];
  __shared__ float Vs[32 * HD];
  __shared__ float Sc[32 * 32];

  const int b = blockIdx.x >> 3;      // N_HEADS == 8
  const int h = blockIdx.x & 7;
  const int tid = threadIdx.x;

  const float* Kp = K  + ((size_t)b * Sk) * D_DIM + h * HD;
  const float* Vp = V  + ((size_t)b * Sk) * D_DIM + h * HD;
  const float* Qp = Qb + (size_t)b * qStride + h * HD;

  for (int i = tid; i < Sk * HD; i += 256) {
    const int r = i >> 6, c = i & 63;
    Ks[i] = Kp[(size_t)r * D_DIM + c];
    Vs[i] = Vp[(size_t)r * D_DIM + c];
  }
  for (int i = tid; i < Sq * HD; i += 256)
    Qs[i] = Qp[(size_t)(i >> 6) * D_DIM + (i & 63)];
  __syncthreads();

  const float scale = 0.04419417382415922f;  // 1/sqrt(512)
  for (int i = tid; i < Sq * Sk; i += 256) {
    const int m = i / Sk, k = i - m * Sk;
    float acc = 0.f;
#pragma unroll 8
    for (int d = 0; d < HD; ++d) acc += Qs[m * HD + d] * Ks[k * HD + d];
    Sc[i] = acc * scale;
  }
  __syncthreads();

  if (tid < Sq) {
    float mx = -INFINITY;
    for (int k = 0; k < Sk; ++k) mx = fmaxf(mx, Sc[tid * Sk + k]);
    float s = 0.f;
    for (int k = 0; k < Sk; ++k) { const float e = __expf(Sc[tid * Sk + k] - mx); Sc[tid * Sk + k] = e; s += e; }
    const float inv = 1.f / s;
    for (int k = 0; k < Sk; ++k) Sc[tid * Sk + k] *= inv;
  }
  __syncthreads();

  for (int i = tid; i < Sq * HD; i += 256) {
    const int m = i >> 6, d = i & 63;
    float acc = 0.f;
    for (int k = 0; k < Sk; ++k) acc += Sc[m * Sk + k] * Vs[k * HD + d];
    Out[((size_t)b * Sq + m) * D_DIM + h * HD + d] = Qs[i] + acc;
  }
}

// ---------------------------------------------------------------------------
// In-place LayerNorm over last dim (512), one block per row.
// Optionally emits a bf16 shadow copy (feeds the next WMMA GEMM's A operand).
// ---------------------------------------------------------------------------
__global__ __launch_bounds__(256)
void layernorm_512(float* __restrict__ X, const float* __restrict__ g,
                   const float* __restrict__ bta, unsigned short* __restrict__ shadow)
{
  __shared__ float r1[256];
  __shared__ float r2[256];
  const int tid = threadIdx.x;
  const size_t rowOff = (size_t)blockIdx.x * D_DIM;
  float* xr = X + rowOff;
  const float x0 = xr[tid];
  const float x1 = xr[tid + 256];
  r1[tid] = x0 + x1;
  r2[tid] = x0 * x0 + x1 * x1;
  __syncthreads();
  for (int off = 128; off > 0; off >>= 1) {
    if (tid < off) { r1[tid] += r1[tid + off]; r2[tid] += r2[tid + off]; }
    __syncthreads();
  }
  const float mean = r1[0] * (1.f / 512.f);
  const float var  = r2[0] * (1.f / 512.f) - mean * mean;
  const float inv  = rsqrtf(var + 1e-5f);
  const float y0 = (x0 - mean) * inv * g[tid]       + bta[tid];
  const float y1 = (x1 - mean) * inv * g[tid + 256] + bta[tid + 256];
  xr[tid]       = y0;
  xr[tid + 256] = y1;
  if (shadow) {
    shadow[rowOff + tid]       = f2bf(y0);
    shadow[rowOff + tid + 256] = f2bf(y1);
  }
}

// ---------------------------------------------------------------------------
// Mean pool over seq dim: out[b,d] = mean_s X[b,s,d]
// ---------------------------------------------------------------------------
__global__ __launch_bounds__(256)
void meanpool_512(const float* __restrict__ X, float* __restrict__ Out)
{
  const int idx = blockIdx.x * 256 + threadIdx.x;   // b*512 + d
  const int b = idx >> 9, d = idx & 511;
  const float* p = X + (size_t)b * S_SEQ * D_DIM + d;
  float acc = 0.f;
#pragma unroll
  for (int s = 0; s < S_SEQ; ++s) acc += p[(size_t)s * D_DIM];
  Out[idx] = acc * (1.f / 32.f);
}

// ---------------------------------------------------------------------------
// Host orchestration
// ---------------------------------------------------------------------------
extern "C" void kernel_launch(void* const* d_in, const int* in_sizes, int n_in,
                              void* d_out, int out_size, void* d_ws, size_t ws_size,
                              hipStream_t stream)
{
  (void)in_sizes; (void)n_in; (void)out_size; (void)ws_size;
  const float* x     = (const float*)d_in[0];
  const float* Ipts  = (const float*)d_in[1];
  const float* W0qkv = (const float*)d_in[2];
  const float* b0qkv = (const float*)d_in[3];
  const float* Wo0   = (const float*)d_in[4];
  const float* bo0   = (const float*)d_in[5];
  const float* g00   = (const float*)d_in[6];
  const float* be00  = (const float*)d_in[7];
  const float* g01   = (const float*)d_in[8];
  const float* be01  = (const float*)d_in[9];
  const float* W1qkv = (const float*)d_in[10];
  const float* b1qkv = (const float*)d_in[11];
  const float* Wo1   = (const float*)d_in[12];
  const float* bo1   = (const float*)d_in[13];
  const float* g10   = (const float*)d_in[14];
  const float* be10  = (const float*)d_in[15];
  const float* g11   = (const float*)d_in[16];
  const float* be11  = (const float*)d_in[17];

  const size_t DD  = (size_t)D_DIM * D_DIM;
  const size_t BIG = (size_t)B_BATCH * S_SEQ * D_DIM;   // 33.55M elements
  const size_t MED = (size_t)B_BATCH * M_IND * D_DIM;   // 16.78M elements
  const int rowsS = B_BATCH * S_SEQ;   // 65536
  const int rowsM = B_BATCH * M_IND;   // 32768

  // ---- workspace carve-up (fp32 buffers, then bf16 shadows/weights) ----
  char* p = (char*)d_ws;
  auto takeF = [&](size_t n) { float* r = (float*)p; p += n * sizeof(float); return r; };
  auto takeH = [&](size_t n) { unsigned short* r = (unsigned short*)p; p += n * sizeof(unsigned short); return r; };
  float* S1 = takeF(BIG);
  float* S2 = takeF(BIG);
  float* S3 = takeF(BIG);          // pass1 out / pass2 in+out
  float* Ha = takeF(MED);
  float* Hb = takeF(MED);
  float* Hc = takeF(MED);
  float* Qp = takeF((size_t)M_IND * D_DIM);
  unsigned short* xb  = takeH(BIG);                 // bf16(x)
  unsigned short* S3b = takeH(BIG);                 // bf16 shadow of h1
  unsigned short* S2b = takeH(BIG);                 // bf16 shadow of ln0(mab1 O)
  unsigned short* Hab = takeH(MED);                 // bf16 shadow of ln0(mab0 O)
  unsigned short* Hbb = takeH(MED);                 // bf16 shadow of Hm
  unsigned short* Ib  = takeH((size_t)64 * D_DIM);  // bf16(I), zero-padded to 64 rows
  unsigned short* Wt[8];
  for (int i = 0; i < 8; ++i) Wt[i] = takeH(DD);    // transposed bf16 weights
  // Wt order: 0:W0q 1:W0k 2:W0v 3:Wo0 4:W1q 5:W1k 6:W1v 7:Wo1

  const dim3 blk(256);
  auto gemm = [&](const unsigned short* A, const unsigned short* W, const float* bias,
                  float* C, int nrows) {
    dim3 grid(D_DIM / 256, (unsigned)((nrows + 63) / 64));
    gemm_bf16_wmma<0><<<grid, blk, 0, stream>>>(A, W, bias, nullptr, C, nrows);
  };
  auto gemmReluRes = [&](const unsigned short* A, const unsigned short* W, const float* bias,
                         const float* Res, float* C, int nrows) {
    dim3 grid(D_DIM / 256, (unsigned)((nrows + 63) / 64));
    gemm_bf16_wmma<1><<<grid, blk, 0, stream>>>(A, W, bias, Res, C, nrows);
  };
  auto lnorm = [&](float* X, const float* g, const float* b, int nrows, unsigned short* sh) {
    layernorm_512<<<dim3((unsigned)nrows), blk, 0, stream>>>(X, g, b, sh);
  };

  // ---- one-time O(N^2) conversions ----
  cvt_bf16<<<dim3((unsigned)(BIG / 256)), blk, 0, stream>>>(x, xb);
  prep_I_bf16<<<dim3(64 * D_DIM / 256), blk, 0, stream>>>(Ipts, Ib);
  wtrans_bf16<<<dim3(DD / 256), blk, 0, stream>>>(W0qkv,          Wt[0]);
  wtrans_bf16<<<dim3(DD / 256), blk, 0, stream>>>(W0qkv + DD,     Wt[1]);
  wtrans_bf16<<<dim3(DD / 256), blk, 0, stream>>>(W0qkv + 2 * DD, Wt[2]);
  wtrans_bf16<<<dim3(DD / 256), blk, 0, stream>>>(Wo0,            Wt[3]);
  wtrans_bf16<<<dim3(DD / 256), blk, 0, stream>>>(W1qkv,          Wt[4]);
  wtrans_bf16<<<dim3(DD / 256), blk, 0, stream>>>(W1qkv + DD,     Wt[5]);
  wtrans_bf16<<<dim3(DD / 256), blk, 0, stream>>>(W1qkv + 2 * DD, Wt[6]);
  wtrans_bf16<<<dim3(DD / 256), blk, 0, stream>>>(Wo1,            Wt[7]);

  // Q-projection of inducing points: batch-invariant, computed once, reused by
  // mab0 in both ISAB passes (also serves as the attention residual term).
  gemm(Ib, Wt[0], b0qkv, Qp, M_IND);

  auto isab = [&](const unsigned short* Xinb, float* Xout, unsigned short* Xoutb) {
    // ---- mab0(I, X) -> Hm [B, M, D] ----
    gemm(Xinb, Wt[1], b0qkv + D_DIM,     S1, rowsS);   // K0
    gemm(Xinb, Wt[2], b0qkv + 2 * D_DIM, S2, rowsS);   // V0
    attn_512<<<dim3(B_BATCH * N_HEADS), blk, 0, stream>>>(Qp, 0L, S1, S2, Ha, M_IND, S_SEQ);
    lnorm(Ha, g00, be00, rowsM, Hab);
    gemmReluRes(Hab, Wt[3], bo0, Ha, Hb, rowsM);
    lnorm(Hb, g01, be01, rowsM, Hbb);                  // Hm = Hb (bf16: Hbb)
    // ---- mab1(X, Hm) -> Xout [B, S, D] ----
    gemm(Xinb, Wt[4], b1qkv,             S1, rowsS);   // Q1 (last read of Xinb)
    gemm(Hbb,  Wt[5], b1qkv + D_DIM,     Ha, rowsM);   // K1
    gemm(Hbb,  Wt[6], b1qkv + 2 * D_DIM, Hc, rowsM);   // V1
    attn_512<<<dim3(B_BATCH * N_HEADS), blk, 0, stream>>>(S1, (long)S_SEQ * D_DIM, Ha, Hc, S2, S_SEQ, M_IND);
    lnorm(S2, g10, be10, rowsS, S2b);
    gemmReluRes(S2b, Wt[7], bo1, S2, Xout, rowsS);
    lnorm(Xout, g11, be11, rowsS, Xoutb);
  };

  isab(xb,  S3, S3b);       // h1 (+ bf16 shadow for pass 2)
  isab(S3b, S3, nullptr);   // h2 (S3b fully consumed before S3 is overwritten)

  meanpool_512<<<dim3((B_BATCH * D_DIM) / 256), blk, 0, stream>>>(S3, (float*)d_out);
}
